// Conversation_Self_Attention_87729001988748
// MI455X (gfx1250) — compile-verified, matrix operand order checked by reading, not run on hardware
//
#include <hip/hip_runtime.h>
#include <hip/hip_bf16.h>

typedef __bf16 bf16_t;
typedef __attribute__((ext_vector_type(16))) __bf16 v16bf;
typedef __attribute__((ext_vector_type(8)))  __bf16 v8bf;
typedef __attribute__((ext_vector_type(8)))  float  v8f;

#define BATCH 64
#define SEQ   512
#define DIM   512
#define NORM  0.03125f  /* 1/sqrt(2*512) = 1/32 exactly */

// ---------------------------------------------------------------------------
// Fragment loader for NT GEMM operands (row-major, K contiguous).
// CDNA5 16-bit A/B fragment: lane L (L<16) holds row L, K = {0..7, 16..23};
// lane L+16 holds row L, K = {8..15, 24..31}.  Two 16-byte loads per lane.
// ---------------------------------------------------------------------------
__device__ __forceinline__ v16bf ld_frag(const bf16_t* __restrict__ rowptr, int koff) {
  v8bf lo = *(const v8bf*)(rowptr + koff);
  v8bf hi = *(const v8bf*)(rowptr + koff + 16);
  return __builtin_shufflevector(lo, hi, 0,1,2,3,4,5,6,7,8,9,10,11,12,13,14,15);
}

__device__ __forceinline__ v16bf ld_frag_lds(const bf16_t* rowptr, int koff) {
  v8bf lo = *(const v8bf*)(rowptr + koff);
  v8bf hi = *(const v8bf*)(rowptr + koff + 16);
  return __builtin_shufflevector(lo, hi, 0,1,2,3,4,5,6,7,8,9,10,11,12,13,14,15);
}

// ---------------------------------------------------------------------------
// f32 -> bf16 bulk convert (8 elems / thread, 16B stores)
// ---------------------------------------------------------------------------
__global__ void __launch_bounds__(256) cvt_f32_bf16(const float* __restrict__ src,
                                                    bf16_t* __restrict__ dst, int n8) {
  int i = blockIdx.x * 256 + threadIdx.x;
  if (i >= n8) return;
  const float4* s = (const float4*)src + (size_t)i * 2;
  float4 a = s[0], b = s[1];
  v8bf o;
  o[0]=(bf16_t)a.x; o[1]=(bf16_t)a.y; o[2]=(bf16_t)a.z; o[3]=(bf16_t)a.w;
  o[4]=(bf16_t)b.x; o[5]=(bf16_t)b.y; o[6]=(bf16_t)b.z; o[7]=(bf16_t)b.w;
  *((v8bf*)dst + i) = o;
}

// ---------------------------------------------------------------------------
// Y[M,N] = A[M,K] * W[N,K]^T + bias[N], bf16 in, bf16 out, f32 accumulate.
// M=32768, N=512, K=512.  Block = 8 waves -> 128x128 tile; wave w owns rows
// m0+16w..+15 across all 8 N sub-tiles.  W tile (128x32) staged in LDS and
// shared by all 8 waves.
// ---------------------------------------------------------------------------
__global__ void __launch_bounds__(256) gemm_nt_bias_bf16(
    const bf16_t* __restrict__ A, const bf16_t* __restrict__ W,
    const float* __restrict__ bias, bf16_t* __restrict__ Y) {
  __shared__ bf16_t Bs[128][40];  // +8 pad elems -> 80B row stride, conflict-free

  const int tid  = threadIdx.x;
  const int lane = tid & 31;
  const int w    = tid >> 5;
  const int nl   = lane & 15;
  const int half = lane >> 4;
  const int m0   = blockIdx.x * 128 + w * 16;
  const int n0   = blockIdx.y * 128;

  const bf16_t* Arow = A + (size_t)(m0 + nl) * DIM;
  const int ldr = tid >> 1;          // 0..127 : W row within tile
  const int ldh = (tid & 1) * 16;    // 16-element half of the 32-wide K slab
  const bf16_t* Wsrc = W + (size_t)(n0 + ldr) * DIM + ldh;

  v8f acc[8];
  const v8f vzero = {0.f,0.f,0.f,0.f,0.f,0.f,0.f,0.f};
  #pragma unroll
  for (int j = 0; j < 8; ++j) acc[j] = vzero;

  for (int k0 = 0; k0 < DIM; k0 += 32) {
    v8bf w0 = *(const v8bf*)(Wsrc + k0);
    v8bf w1 = *(const v8bf*)(Wsrc + k0 + 8);
    __syncthreads();                      // prior iteration done reading Bs
    *(v8bf*)&Bs[ldr][ldh]     = w0;
    *(v8bf*)&Bs[ldr][ldh + 8] = w1;
    __syncthreads();
    v16bf afrag = ld_frag(Arow, k0 + half * 8);
    #pragma unroll
    for (int j = 0; j < 8; ++j) {
      v16bf bfrag = ld_frag_lds(&Bs[j * 16 + nl][0], half * 8);
      acc[j] = __builtin_amdgcn_wmma_f32_16x16x32_bf16(
          false, afrag, false, bfrag, (short)0, acc[j], false, false);
    }
  }

  const int mb = half * 8;
  #pragma unroll
  for (int j = 0; j < 8; ++j) {
    const int col = n0 + j * 16 + nl;
    const float bv = bias[col];
    #pragma unroll
    for (int r = 0; r < 8; ++r)
      Y[(size_t)(m0 + mb + r) * DIM + col] = (bf16_t)(acc[j][r] + bv);
  }
}

// ---------------------------------------------------------------------------
// Fused scores + softmax.  Block handles 16 query rows x all 512 keys of one
// batch: acc = Q.K^T + Uq.Uk^T (two WMMAs per tile per K-step), scale, +bias,
// scores -> LDS, 16-lane shuffle row reduction, exp, normalized f32 att to
// d_out and bf16 att to workspace.
// ---------------------------------------------------------------------------
__global__ void __launch_bounds__(256) attn_score_softmax(
    const bf16_t* __restrict__ Q,  const bf16_t* __restrict__ K,
    const bf16_t* __restrict__ Uq, const bf16_t* __restrict__ Uk,
    const float* __restrict__ bias,
    float* __restrict__ att_out, bf16_t* __restrict__ attB) {
  __shared__ float sc[16][SEQ + 4];

  const int tid  = threadIdx.x;
  const int lane = tid & 31;
  const int w    = tid >> 5;
  const int nl   = lane & 15;
  const int half = lane >> 4;
  const int b    = blockIdx.y;
  const int i0   = blockIdx.x * 16;
  const size_t bo = (size_t)b * SEQ * DIM;

  const bf16_t* Qr  = Q  + bo + (size_t)(i0 + nl) * DIM;
  const bf16_t* Uqr = Uq + bo + (size_t)(i0 + nl) * DIM;
  const bf16_t* Kb  = K  + bo;
  const bf16_t* Ukb = Uk + bo;

  v8f acc[4];
  const v8f vzero = {0.f,0.f,0.f,0.f,0.f,0.f,0.f,0.f};
  #pragma unroll
  for (int q = 0; q < 4; ++q) acc[q] = vzero;

  for (int k0 = 0; k0 < DIM; k0 += 32) {
    const int ko = k0 + half * 8;
    v16bf aQ = ld_frag(Qr,  ko);
    v16bf aU = ld_frag(Uqr, ko);
    #pragma unroll
    for (int q = 0; q < 4; ++q) {
      const int nt = (w + q * 8) * 16;
      v16bf bK = ld_frag(Kb  + (size_t)(nt + nl) * DIM, ko);
      v16bf bU = ld_frag(Ukb + (size_t)(nt + nl) * DIM, ko);
      acc[q] = __builtin_amdgcn_wmma_f32_16x16x32_bf16(
          false, aQ, false, bK, (short)0, acc[q], false, false);
      acc[q] = __builtin_amdgcn_wmma_f32_16x16x32_bf16(
          false, aU, false, bU, (short)0, acc[q], false, false);
    }
  }

  const int mb = half * 8;
  #pragma unroll
  for (int q = 0; q < 4; ++q) {
    const int col = (w + q * 8) * 16 + nl;
    const float bv = bias[col];
    #pragma unroll
    for (int r = 0; r < 8; ++r)
      sc[mb + r][col] = acc[q][r] * NORM + bv;
  }
  __syncthreads();

  // softmax: 16 threads per row (lanes 0..15 / 16..31 of a wave = one row each)
  const int row = tid >> 4;
  const int g   = tid & 15;
  float mx = -3.0e38f;
  #pragma unroll 8
  for (int c = 0; c < 32; ++c) mx = fmaxf(mx, sc[row][g * 32 + c]);
  mx = fmaxf(mx, __shfl_xor(mx, 1, 32));
  mx = fmaxf(mx, __shfl_xor(mx, 2, 32));
  mx = fmaxf(mx, __shfl_xor(mx, 4, 32));
  mx = fmaxf(mx, __shfl_xor(mx, 8, 32));
  float sum = 0.f;
  #pragma unroll 8
  for (int c = 0; c < 32; ++c) {
    float e = __expf(sc[row][g * 32 + c] - mx);
    sc[row][g * 32 + c] = e;
    sum += e;
  }
  sum += __shfl_xor(sum, 1, 32);
  sum += __shfl_xor(sum, 2, 32);
  sum += __shfl_xor(sum, 4, 32);
  sum += __shfl_xor(sum, 8, 32);
  const float inv = 1.0f / sum;
  const size_t ob = (size_t)b * SEQ * SEQ + (size_t)(i0 + row) * SEQ;
  #pragma unroll 8
  for (int c = 0; c < 32; ++c) {
    const int col = g * 32 + c;
    float a = sc[row][col] * inv;
    att_out[ob + col] = a;
    attB[ob + col] = (bf16_t)a;
  }
}

// ---------------------------------------------------------------------------
// out[b,i,j] = sum_t att[b,i,t] * V[b,j,t]   (batched NT GEMM, f32 out)
// ---------------------------------------------------------------------------
__global__ void __launch_bounds__(256) gemm_att_v(
    const bf16_t* __restrict__ Att, const bf16_t* __restrict__ V,
    float* __restrict__ Out) {
  __shared__ bf16_t Bs[128][40];

  const int tid  = threadIdx.x;
  const int lane = tid & 31;
  const int w    = tid >> 5;
  const int nl   = lane & 15;
  const int half = lane >> 4;
  const int b    = blockIdx.z;
  const int m0   = blockIdx.x * 128 + w * 16;
  const int n0   = blockIdx.y * 128;

  const bf16_t* Arow = Att + (size_t)b * SEQ * SEQ + (size_t)(m0 + nl) * SEQ;
  const int ldr = tid >> 1;
  const int ldh = (tid & 1) * 16;
  const bf16_t* Vsrc = V + (size_t)b * SEQ * DIM + (size_t)(n0 + ldr) * DIM + ldh;

  v8f acc[8];
  const v8f vzero = {0.f,0.f,0.f,0.f,0.f,0.f,0.f,0.f};
  #pragma unroll
  for (int j = 0; j < 8; ++j) acc[j] = vzero;

  for (int k0 = 0; k0 < SEQ; k0 += 32) {
    v8bf w0 = *(const v8bf*)(Vsrc + k0);
    v8bf w1 = *(const v8bf*)(Vsrc + k0 + 8);
    __syncthreads();
    *(v8bf*)&Bs[ldr][ldh]     = w0;
    *(v8bf*)&Bs[ldr][ldh + 8] = w1;
    __syncthreads();
    v16bf afrag = ld_frag(Arow, k0 + half * 8);
    #pragma unroll
    for (int j = 0; j < 8; ++j) {
      v16bf bfrag = ld_frag_lds(&Bs[j * 16 + nl][0], half * 8);
      acc[j] = __builtin_amdgcn_wmma_f32_16x16x32_bf16(
          false, afrag, false, bfrag, (short)0, acc[j], false, false);
    }
  }

  float* outb = Out + (size_t)b * SEQ * SEQ;
  const int mb = half * 8;
  #pragma unroll
  for (int j = 0; j < 8; ++j) {
    const int col = n0 + j * 16 + nl;
    #pragma unroll
    for (int r = 0; r < 8; ++r)
      outb[(size_t)(m0 + mb + r) * SEQ + col] = acc[j][r];
  }
}

// ---------------------------------------------------------------------------
extern "C" void kernel_launch(void* const* d_in, const int* in_sizes, int n_in,
                              void* d_out, int out_size, void* d_ws, size_t ws_size,
                              hipStream_t stream) {
  (void)in_sizes; (void)n_in; (void)out_size; (void)ws_size;

  const float* sent = (const float*)d_in[0];
  const float* pos  = (const float*)d_in[1];
  // d_in[2] = branch_emb (unused by reference)
  const float* Wq   = (const float*)d_in[3];
  const float* bq   = (const float*)d_in[4];
  const float* Wk   = (const float*)d_in[5];
  const float* bk   = (const float*)d_in[6];
  const float* Wv   = (const float*)d_in[7];
  const float* bv   = (const float*)d_in[8];
  const float* Wpq  = (const float*)d_in[9];
  const float* bpq  = (const float*)d_in[10];
  const float* Wpk  = (const float*)d_in[11];
  const float* bpk  = (const float*)d_in[12];
  const float* bias = (const float*)d_in[13];

  float* out_att = (float*)d_out;
  float* out_bmm = out_att + (size_t)BATCH * SEQ * SEQ;

  constexpr size_t EMB = (size_t)BATCH * SEQ * DIM;  // 16,777,216
  constexpr size_t WSZ = (size_t)DIM * DIM;          //    262,144

  bf16_t* sentB = (bf16_t*)d_ws;
  bf16_t* posB  = sentB + EMB;
  bf16_t* Wqb   = posB  + EMB;
  bf16_t* Wkb   = Wqb + WSZ;
  bf16_t* Wvb   = Wkb + WSZ;
  bf16_t* Wpqb  = Wvb + WSZ;
  bf16_t* Wpkb  = Wpqb + WSZ;
  bf16_t* Qb    = Wpkb + WSZ;
  bf16_t* Kb    = Qb  + EMB;
  bf16_t* Vb    = Kb  + EMB;
  bf16_t* Uqb   = Vb  + EMB;
  bf16_t* Ukb   = Uqb + EMB;
  bf16_t* attB  = Ukb + EMB;

  // 1) f32 -> bf16 conversions
  cvt_f32_bf16<<<(unsigned)(EMB / 8 / 256), 256, 0, stream>>>(sent, sentB, (int)(EMB / 8));
  cvt_f32_bf16<<<(unsigned)(EMB / 8 / 256), 256, 0, stream>>>(pos,  posB,  (int)(EMB / 8));
  cvt_f32_bf16<<<(unsigned)(WSZ / 8 / 256), 256, 0, stream>>>(Wq,  Wqb,  (int)(WSZ / 8));
  cvt_f32_bf16<<<(unsigned)(WSZ / 8 / 256), 256, 0, stream>>>(Wk,  Wkb,  (int)(WSZ / 8));
  cvt_f32_bf16<<<(unsigned)(WSZ / 8 / 256), 256, 0, stream>>>(Wv,  Wvb,  (int)(WSZ / 8));
  cvt_f32_bf16<<<(unsigned)(WSZ / 8 / 256), 256, 0, stream>>>(Wpq, Wpqb, (int)(WSZ / 8));
  cvt_f32_bf16<<<(unsigned)(WSZ / 8 / 256), 256, 0, stream>>>(Wpk, Wpkb, (int)(WSZ / 8));

  // 2) projections (WMMA bf16, f32 accumulate)
  dim3 gp(256, 4);
  gemm_nt_bias_bf16<<<gp, 256, 0, stream>>>(sentB, Wqb,  bq,  Qb);
  gemm_nt_bias_bf16<<<gp, 256, 0, stream>>>(sentB, Wkb,  bk,  Kb);
  gemm_nt_bias_bf16<<<gp, 256, 0, stream>>>(sentB, Wvb,  bv,  Vb);
  gemm_nt_bias_bf16<<<gp, 256, 0, stream>>>(posB,  Wpqb, bpq, Uqb);
  gemm_nt_bias_bf16<<<gp, 256, 0, stream>>>(posB,  Wpkb, bpk, Ukb);

  // 3) fused scores + softmax
  dim3 ga(32, 64);
  attn_score_softmax<<<ga, 256, 0, stream>>>(Qb, Kb, Uqb, Ukb, bias, out_att, attB);

  // 4) output bmm: att * V^T
  dim3 go(4, 4, 64);
  gemm_att_v<<<go, 256, 0, stream>>>(attB, Vb, out_bmm);
}
